// SelfAttention_12833362280628
// MI455X (gfx1250) — matrix operand order repaired and checked by reading, not compiled, not run on hardware
//
#include <hip/hip_runtime.h>
#include <hip/hip_bf16.h>

// ---------------------------------------------------------------------------
// Problem constants (from the reference)
// ---------------------------------------------------------------------------
#define NNODE   512            // N
#define XD      256            // XDIM (= N_HEAD * DF = 8*32)
#define OUTROW  (NNODE * XD)   // 131072 floats per query row i
// scale applied inside exp2: log2(e) / sqrt(DF)
#define QSCALE  (1.4426950408889634f / 5.656854249492380f)

typedef __attribute__((ext_vector_type(16))) _Float16 v16h;
typedef __attribute__((ext_vector_type(8)))  float    v8f;

#if __has_builtin(__builtin_amdgcn_exp2f)
#define FAST_EXP2(x) __builtin_amdgcn_exp2f(x)
#else
#define FAST_EXP2(x) exp2f(x)
#endif
#if __has_builtin(__builtin_amdgcn_rcpf)
#define FAST_RCP(x) __builtin_amdgcn_rcpf(x)
#else
#define FAST_RCP(x) (1.0f / (x))
#endif

// ---------------------------------------------------------------------------
// Kernel 1: projection GEMM  out[512,256] = x[512,256] @ W[256,256] + b[256]
// One wave (32 threads) computes one 16x16 output tile with
// v_wmma_f32_16x16x32_f16, looping K=256 in 8 chunks of 32.
//
// Fragment layouts (CDNA5 ISA 7.12.2, wave32):
//   A 16x32 f16 : lane l -> row m = l&15 ; kgrp = l>>4 ;
//                 halves 0..7  = k = 32*kk + kgrp*8 + h
//                 halves 8..15 = k = 32*kk + 16 + kgrp*8 + (h-8)
//   B 32x16 f16 : lane l -> col n = l&15 ; same k mapping as A (kgrp*8 + h, 16 + kgrp*8 + (h-8))
//   C/D 16x16 f32: vgpr v, lane l -> m = v + (l>>4)*8 , n = l&15
// ---------------------------------------------------------------------------
__global__ __launch_bounds__(32)
void proj_gemm_wmma(const float* __restrict__ x,
                    const float* __restrict__ W,
                    const float* __restrict__ bias,
                    float* __restrict__ out)
{
    const int lane = threadIdx.x;          // 0..31, one wave per block
    const int l15  = lane & 15;
    const int lhi  = lane >> 4;

    const int tm = blockIdx.y * 16;        // output row tile
    const int tn = blockIdx.x * 16;        // output col tile

    const int m = tm + l15;                // A row this lane feeds
    const int n = tn + l15;                // B col this lane feeds

    v8f acc = {};

#pragma unroll
    for (int kk = 0; kk < XD / 32; ++kk) {
        const int k0 = kk * 32;

        v16h a, b;
        // A fragment: two contiguous runs of 8 floats from row m of x
        const float* xa = x + (size_t)m * XD + k0 + lhi * 8;
#pragma unroll
        for (int h = 0; h < 8; ++h) a[h]     = (_Float16)xa[h];
#pragma unroll
        for (int h = 0; h < 8; ++h) a[8 + h] = (_Float16)xa[16 + h];

        // B fragment: 16 rows of W, fixed column n (coalesced across lanes)
        const float* wb = W + (size_t)(k0 + lhi * 8) * XD + n;
#pragma unroll
        for (int h = 0; h < 16; ++h) b[h] = (_Float16)wb[(size_t)(h + (h >> 3) * 8) * XD];

        acc = __builtin_amdgcn_wmma_f32_16x16x32_f16(
            /*neg_a=*/false, a, /*neg_b=*/false, b,
            /*c_mod=*/(short)0, acc, /*reuse_a=*/false, /*reuse_b=*/false);
    }

    const float bn = bias[n];
#pragma unroll
    for (int v = 0; v < 8; ++v) {
        const int mo = tm + v + lhi * 8;
        out[(size_t)mo * XD + n] = acc[v] + bn;
    }
}

// ---------------------------------------------------------------------------
// Kernel 2: attn[i, j, c] = softmax_j( Q[i,c] * K[j,c] / sqrt(32) )
// Block owns 4 query rows (i0..i0+3) and all 256 channels (thread t -> c=t).
// Pass 1 accumulates sum_j exp2(qs*k); pass 2 recomputes and writes.
// K rows are contiguous 1KB loads (coalesced, L2-resident); output rows are
// contiguous 1KB non-temporal stores.
// ---------------------------------------------------------------------------
#define IPB 4   // query rows per block

__global__ __launch_bounds__(XD)
void attn_softmax(const float* __restrict__ Q,
                  const float* __restrict__ K,
                  float* __restrict__ out)
{
    const int c  = threadIdx.x;            // channel 0..255
    const int i0 = blockIdx.x * IPB;

    float qs[IPB];
#pragma unroll
    for (int r = 0; r < IPB; ++r)
        qs[r] = Q[(size_t)(i0 + r) * XD + c] * QSCALE;

    // ---- pass 1: denominators ----
    float sum[IPB];
#pragma unroll
    for (int r = 0; r < IPB; ++r) sum[r] = 0.0f;

#pragma unroll 4
    for (int j = 0; j < NNODE; ++j) {
        const float k = K[(size_t)j * XD + c];
#pragma unroll
        for (int r = 0; r < IPB; ++r)
            sum[r] += FAST_EXP2(qs[r] * k);
    }

    float inv[IPB];
#pragma unroll
    for (int r = 0; r < IPB; ++r) inv[r] = FAST_RCP(sum[r]);

    // ---- pass 2: recompute exp, normalize, stream out ----
#pragma unroll 4
    for (int j = 0; j < NNODE; ++j) {
        const float k = K[(size_t)j * XD + c];
#pragma unroll
        for (int r = 0; r < IPB; ++r) {
            const float e = FAST_EXP2(qs[r] * k) * inv[r];
            __builtin_nontemporal_store(
                e, out + (size_t)(i0 + r) * OUTROW + (size_t)j * XD + c);
        }
    }
}

// ---------------------------------------------------------------------------
// Launcher
// ---------------------------------------------------------------------------
extern "C" void kernel_launch(void* const* d_in, const int* in_sizes, int n_in,
                              void* d_out, int out_size, void* d_ws, size_t ws_size,
                              hipStream_t stream) {
    const float* x  = (const float*)d_in[0];
    const float* Wq = (const float*)d_in[1];
    const float* bq = (const float*)d_in[2];
    const float* Wk = (const float*)d_in[3];
    const float* bk = (const float*)d_in[4];
    // d_in[5] (Wv), d_in[6] (bv) are dead code in the reference output.

    float* Qws = (float*)d_ws;                       // 512*256 floats
    float* Kws = Qws + (size_t)NNODE * XD;           // 512*256 floats

    dim3 gg(XD / 16, NNODE / 16);                    // 16 x 32 tiles
    proj_gemm_wmma<<<gg, 32, 0, stream>>>(x, Wq, bq, Qws);
    proj_gemm_wmma<<<gg, 32, 0, stream>>>(x, Wk, bk, Kws);

    attn_softmax<<<NNODE / IPB, XD, 0, stream>>>(Qws, Kws, (float*)d_out);
}
